// GeniepathLazyEncoder_59871844106896
// MI455X (gfx1250) — compile-verified, matrix operand order checked by reading, not run on hardware
//
#include <hip/hip_runtime.h>
#include <hip/hip_bf16.h>
#include <math.h>

// ---------------------------------------------------------------------------
// GeniePath-style encoder for MI455X (gfx1250, wave32).
//  - All dense math on v_wmma_f32_16x16x32_bf16 (bf16 operands, f32 accum).
//  - Weights pre-packed to bf16 [N,K] (B-transposed) once per call.
//  - Activations produced directly in bf16 by producer kernels.
//  - Full LSTM step fused into one kernel (3 GEMMs in registers, 4 gate
//    accumulators, recurrence applied on the WMMA fragments).
//  - Edge softmax/aggregation: L2-resident float atomics.
// ---------------------------------------------------------------------------

static constexpr int HDIM  = 128;   // hidden size H
static constexpr int FDIM  = 256;   // input feature size F
static constexpr int NHOPS = 3;
static constexpr int TPB   = 256;   // 8 waves per block

typedef __attribute__((ext_vector_type(16))) __bf16 bf16x16;
typedef __attribute__((ext_vector_type(8)))  float  f32x8;

// --- WMMA fragment loader --------------------------------------------------
// 16-bit A-matrix 16x32 layout (ISA 7.12.2): lane L supplies row M=L&15;
// K base = (L>>4)*8; frag elems 0..7 -> K+0..7, elems 8..15 -> K+16..23.
// B fragment mirrors A with the B-column as the "row" (we pre-pack B as [N,K]).
__device__ __forceinline__ bf16x16 load_frag(const __bf16* __restrict__ p) {
  bf16x16 f;
#pragma unroll
  for (int j = 0; j < 8; ++j) f[j] = p[j];
#pragma unroll
  for (int j = 0; j < 8; ++j) f[8 + j] = p[16 + j];
  return f;
}

#define WMMA_BF16(a, b, c) \
  __builtin_amdgcn_wmma_f32_16x16x32_bf16(false, (a), false, (b), (short)0, (c), false, false)

__device__ __forceinline__ float sigmoidf_(float x) { return 1.f / (1.f + __expf(-x)); }

// --- blocked GEMM: C[M,Nn] = A[M,K](bf16) * B^T (B packed [Nn,K] bf16) ------
// Wave computes a 16x64 block: 1 A fragment feeds 4 B tiles -> 4 WMMAs/K-step.
__global__ void wmma_gemm_bt(const __bf16* __restrict__ A, const __bf16* __restrict__ B,
                             const float* __restrict__ bias,
                             float* __restrict__ Cf, __bf16* __restrict__ Cbf,
                             int M, int Nn, int K, int ldb)
{
  const int wave    = (blockIdx.x * blockDim.x + threadIdx.x) >> 5;
  const int lane    = threadIdx.x & 31;
  const int tilesN4 = Nn >> 6;                  // 64-column groups
  if (wave >= (M >> 4) * tilesN4) return;
  const int tm   = wave / tilesN4;
  const int tq   = wave - tm * tilesN4;
  const int half = lane >> 4, r = lane & 15;
  const int arow = tm * 16 + r;
  const int col0 = tq * 64 + r;                 // tile t covers col0 + t*16

  f32x8 acc0 = {}, acc1 = {}, acc2 = {}, acc3 = {};
  for (int k0 = 0; k0 < K; k0 += 32) {
    const int kb = k0 + half * 8;
    bf16x16 a  = load_frag(A + (size_t)arow * K + kb);
    bf16x16 b0 = load_frag(B + (size_t)(col0     ) * ldb + kb);
    bf16x16 b1 = load_frag(B + (size_t)(col0 + 16) * ldb + kb);
    bf16x16 b2 = load_frag(B + (size_t)(col0 + 32) * ldb + kb);
    bf16x16 b3 = load_frag(B + (size_t)(col0 + 48) * ldb + kb);
    acc0 = WMMA_BF16(a, b0, acc0);
    acc1 = WMMA_BF16(a, b1, acc1);
    acc2 = WMMA_BF16(a, b2, acc2);
    acc3 = WMMA_BF16(a, b3, acc3);
  }

  // 32-bit C/D 16x16 layout: lane r = col, VGPR j -> row half*8 + j
  const int m0 = tm * 16 + half * 8;
  const f32x8 accs[4] = {acc0, acc1, acc2, acc3};
#pragma unroll
  for (int t = 0; t < 4; ++t) {
    const int col = col0 + t * 16;
    const float bv = bias ? bias[col] : 0.f;
#pragma unroll
    for (int j = 0; j < 8; ++j) {
      const float v = accs[t][j] + bv;
      const size_t idx = (size_t)(m0 + j) * Nn + col;
      if (Cf)  Cf[idx]  = v;
      if (Cbf) Cbf[idx] = (__bf16)v;
    }
  }
}

// --- fused LSTM step --------------------------------------------------------
// gates = Ht @ Wih[:, :H]^T + Xc @ Wih[:, H:]^T + Hp @ Whh^T   (torch i,f,g,o)
// Wave owns 16 rows x 16 hidden cols; 4 accumulators = the 4 gate blocks.
// Applies recurrence on fragments: c (f32, in-place, wave-exclusive),
// Hn (bf16, ping-pong buffer != Hp).
__global__ void lstm_fused(const __bf16* __restrict__ Ht, const __bf16* __restrict__ Xc,
                           const __bf16* __restrict__ Hp,
                           const __bf16* __restrict__ Wih, const __bf16* __restrict__ Whh,
                           float* __restrict__ c, __bf16* __restrict__ Hn, int M)
{
  const int wave   = (blockIdx.x * blockDim.x + threadIdx.x) >> 5;
  const int lane   = threadIdx.x & 31;
  const int tilesC = HDIM >> 4;                 // 8 hidden-col tiles
  if (wave >= (M >> 4) * tilesC) return;
  const int tm   = wave / tilesC;
  const int tc   = wave - tm * tilesC;
  const int half = lane >> 4, r = lane & 15;
  const int arow = tm * 16 + r;

  f32x8 acc[4] = {{}, {}, {}, {}};
  for (int k0 = 0; k0 < HDIM; k0 += 32) {
    const int kb = k0 + half * 8;
    bf16x16 a1 = load_frag(Ht + (size_t)arow * HDIM + kb);
    bf16x16 a2 = load_frag(Xc + (size_t)arow * HDIM + kb);
    bf16x16 a3 = load_frag(Hp + (size_t)arow * HDIM + kb);
#pragma unroll
    for (int g = 0; g < 4; ++g) {
      const int brow = g * HDIM + tc * 16 + r;  // row of W (output-gate dim 4H)
      bf16x16 b1 = load_frag(Wih + (size_t)brow * (2 * HDIM) + kb);
      bf16x16 b2 = load_frag(Wih + (size_t)brow * (2 * HDIM) + HDIM + kb);
      bf16x16 b3 = load_frag(Whh + (size_t)brow * HDIM + kb);
      acc[g] = WMMA_BF16(a1, b1, acc[g]);
      acc[g] = WMMA_BF16(a2, b2, acc[g]);
      acc[g] = WMMA_BF16(a3, b3, acc[g]);
    }
  }

  const int m0  = tm * 16 + half * 8;
  const int col = tc * 16 + r;
#pragma unroll
  for (int j = 0; j < 8; ++j) {
    const size_t idx = (size_t)(m0 + j) * HDIM + col;
    const float ig = sigmoidf_(acc[0][j]);
    const float fg = sigmoidf_(acc[1][j]);
    const float gg = tanhf   (acc[2][j]);
    const float og = sigmoidf_(acc[3][j]);
    const float cn = fg * c[idx] + ig * gg;
    c[idx]  = cn;
    Hn[idx] = (__bf16)(og * tanhf(cn));
  }
}

// --- weight/activation packing ---------------------------------------------

__global__ void cvt_bf16_kernel(const float* __restrict__ s, __bf16* __restrict__ d, int n) {
  const int i = blockIdx.x * blockDim.x + threadIdx.x;
  if (i < n) d[i] = (__bf16)s[i];
}

// s: [K, Nn] row-major  ->  d: [Nn, K]  (B-transposed pack)
__global__ void pack_transpose(const float* __restrict__ s, __bf16* __restrict__ d,
                               int Nn, int K) {
  const int i = blockIdx.x * blockDim.x + threadIdx.x;
  if (i >= Nn * K) return;
  const int n = i / K, k = i - n * K;
  d[i] = (__bf16)s[(size_t)k * Nn + n];
}

// --- GAT edge phase ---------------------------------------------------------

__device__ __forceinline__ float leaky02(float x) { return x > 0.f ? x : 0.2f * x; }

__device__ __forceinline__ void atomic_max_f32(float* addr, float val) {
  if (val >= 0.f) atomicMax((int*)addr, __float_as_int(val));
  else            atomicMin((unsigned int*)addr, __float_as_uint(val));
}

__global__ void attn_scores(const float* __restrict__ xp,
                            const float* __restrict__ a_src,
                            const float* __restrict__ a_dst,
                            float* __restrict__ e_src, float* __restrict__ e_dst, int n)
{
  const int wave = (blockIdx.x * blockDim.x + threadIdx.x) >> 5;
  const int lane = threadIdx.x & 31;
  if (wave >= n) return;
  const float* row = xp + (size_t)wave * HDIM;
  float s = 0.f, d = 0.f;
#pragma unroll
  for (int k = lane; k < HDIM; k += 32) {
    const float v = row[k];
    s += v * a_src[k];
    d += v * a_dst[k];
  }
#pragma unroll
  for (int off = 16; off > 0; off >>= 1) {
    s += __shfl_xor(s, off, 32);
    d += __shfl_xor(d, off, 32);
  }
  if (lane == 0) { e_src[wave] = s; e_dst[wave] = d; }
}

__global__ void gat_init(float* __restrict__ m, float* __restrict__ denom, int n) {
  const int i = blockIdx.x * blockDim.x + threadIdx.x;
  if (i < n) { m[i] = -INFINITY; denom[i] = 0.f; }
}

__global__ void edge_max(const int* __restrict__ src, const int* __restrict__ dst,
                         const float* __restrict__ es, const float* __restrict__ ed,
                         float* __restrict__ m, int e)
{
  const int i = blockIdx.x * blockDim.x + threadIdx.x;
  if (i >= e) return;
  const int s = src[i], d = dst[i];
  atomic_max_f32(&m[d], leaky02(es[s] + ed[d]));
}

__global__ void m_fixup(float* __restrict__ m, int n) {
  const int i = blockIdx.x * blockDim.x + threadIdx.x;
  if (i < n && !isfinite(m[i])) m[i] = 0.f;   // empty segments -> safe
}

__global__ void edge_exp(const int* __restrict__ src, const int* __restrict__ dst,
                         const float* __restrict__ es, const float* __restrict__ ed,
                         const float* __restrict__ m, const float* __restrict__ xp,
                         float* __restrict__ exb, float* __restrict__ denom, int e)
{
  const int i = blockIdx.x * blockDim.x + threadIdx.x;
  if (i >= e) return;
  const int s = src[i], d = dst[i];
  __builtin_prefetch(xp + (size_t)s * HDIM, 0, 1);   // warm L2 for scatter pass
  const float x = __expf(leaky02(es[s] + ed[d]) - m[d]);
  exb[i] = x;
  atomicAdd(&denom[d], x);
}

__global__ void edge_scatter(const int* __restrict__ src, const int* __restrict__ dst,
                             const float* __restrict__ exb, const float* __restrict__ denom,
                             const float* __restrict__ xp, float* __restrict__ agg, int e)
{
  const int t    = blockIdx.x * blockDim.x + threadIdx.x;
  const int i    = t >> 5;
  const int lane = t & 31;
  if (i >= e) return;
  const int s = src[i], d = dst[i];
  const float coef = exb[i] / (denom[d] + 1e-16f);
  const float4 v = ((const float4*)(xp + (size_t)s * HDIM))[lane];
  float* o = agg + (size_t)d * HDIM + lane * 4;
  atomicAdd(o + 0, coef * v.x);
  atomicAdd(o + 1, coef * v.y);
  atomicAdd(o + 2, coef * v.z);
  atomicAdd(o + 3, coef * v.w);
}

__global__ void gat_finish(const float* __restrict__ agg, const float* __restrict__ b,
                           __bf16* __restrict__ ht, int total)
{
  const int i = blockIdx.x * blockDim.x + threadIdx.x;
  if (i < total) ht[i] = (__bf16)tanhf(agg[i] + b[i & (HDIM - 1)]);
}

// ---------------------------------------------------------------------------

extern "C" void kernel_launch(void* const* d_in, const int* in_sizes, int n_in,
                              void* d_out, int out_size, void* d_ws, size_t ws_size,
                              hipStream_t stream)
{
  const float* x     = (const float*)d_in[0];
  const int*   eidx  = (const int*)d_in[1];
  const float* w_in  = (const float*)d_in[2];
  const float* b_in  = (const float*)d_in[3];
  const float* w_gat = (const float*)d_in[4];
  const float* att_s = (const float*)d_in[5];
  const float* att_d = (const float*)d_in[6];
  const float* b_gat = (const float*)d_in[7];
  const float* w_ih  = (const float*)d_in[8];
  const float* w_hh  = (const float*)d_in[9];
  const float* w_out = (const float*)d_in[10];
  const float* b_out = (const float*)d_in[11];
  float* out = (float*)d_out;

  const int N = in_sizes[0] / FDIM;        // 100000 (multiple of 16)
  const int E = in_sizes[1] / 2;           // 1,600,000
  const int* src = eidx;
  const int* dst = eidx + E;

  // --- workspace carve-up (256B aligned) ---
  char* p = (char*)d_ws;
  auto allocB = [&](size_t bytes) {
    void* q = (void*)p;
    p += ((bytes + 255) & ~(size_t)255);
    return q;
  };
  __bf16* x_bf    = (__bf16*)allocB((size_t)N * FDIM * 2);
  __bf16* h0_bf   = (__bf16*)allocB((size_t)N * HDIM * 2);
  __bf16* htmp_bf = (__bf16*)allocB((size_t)N * HDIM * 2);
  __bf16* hbufA   = (__bf16*)allocB((size_t)N * HDIM * 2);
  __bf16* hbufB   = (__bf16*)allocB((size_t)N * HDIM * 2);
  float*  xp      = (float*) allocB((size_t)N * HDIM * 4);
  float*  agg     = (float*) allocB((size_t)N * HDIM * 4);
  float*  cbuf    = (float*) allocB((size_t)N * HDIM * 4);
  float*  es      = (float*) allocB((size_t)N * 4);
  float*  ed      = (float*) allocB((size_t)N * 4);
  float*  mbuf    = (float*) allocB((size_t)N * 4);
  float*  den     = (float*) allocB((size_t)N * 4);
  float*  exb     = (float*) allocB((size_t)E * 4);
  __bf16* w_inT   = (__bf16*)allocB((size_t)HDIM * FDIM * 2);           // [H,F]
  __bf16* w_gatT  = (__bf16*)allocB((size_t)NHOPS * HDIM * HDIM * 2);   // [H,H] x hops
  __bf16* w_outT  = (__bf16*)allocB((size_t)HDIM * HDIM * 2);           // [OUT,H]
  __bf16* wih_bf  = (__bf16*)allocB((size_t)NHOPS * 4 * HDIM * 2 * HDIM * 2);
  __bf16* whh_bf  = (__bf16*)allocB((size_t)NHOPS * 4 * HDIM * HDIM * 2);
  (void)ws_size; (void)n_in; (void)out_size;

  auto wgrid = [&](long waves) { return (int)((waves * 32 + TPB - 1) / TPB); };
  auto egrid = [&](long n)     { return (int)((n + TPB - 1) / TPB); };

  // --- pack weights + input to bf16 ---
  cvt_bf16_kernel<<<egrid((long)N * FDIM), TPB, 0, stream>>>(x, x_bf, N * FDIM);
  pack_transpose<<<egrid((long)HDIM * FDIM), TPB, 0, stream>>>(w_in, w_inT, HDIM, FDIM);
  for (int hop = 0; hop < NHOPS; ++hop)
    pack_transpose<<<egrid((long)HDIM * HDIM), TPB, 0, stream>>>(
        w_gat + (size_t)hop * HDIM * HDIM, w_gatT + (size_t)hop * HDIM * HDIM, HDIM, HDIM);
  pack_transpose<<<egrid((long)HDIM * HDIM), TPB, 0, stream>>>(w_out, w_outT, HDIM, HDIM);
  cvt_bf16_kernel<<<egrid((long)NHOPS * 4 * HDIM * 2 * HDIM), TPB, 0, stream>>>(
      w_ih, wih_bf, NHOPS * 4 * HDIM * 2 * HDIM);   // already [4H,2H] rows = B^T
  cvt_bf16_kernel<<<egrid((long)NHOPS * 4 * HDIM * HDIM), TPB, 0, stream>>>(
      w_hh, whh_bf, NHOPS * 4 * HDIM * HDIM);       // already [4H,H] rows = B^T

  hipMemsetAsync(hbufA, 0, (size_t)N * HDIM * sizeof(__bf16), stream);  // h = 0
  hipMemsetAsync(cbuf,  0, (size_t)N * HDIM * sizeof(float),  stream);  // c = 0

  // h0 = x @ w_in + b_in  (bf16 out; h0 is only ever a GEMM A-operand)
  wmma_gemm_bt<<<wgrid((long)(N / 16) * (HDIM / 64)), TPB, 0, stream>>>(
      x_bf, w_inT, b_in, nullptr, h0_bf, N, HDIM, FDIM, FDIM);

  __bf16* hp = hbufA;   // previous hidden state (zeros for hop 0)
  __bf16* hn = hbufB;   // next hidden state

  for (int hop = 0; hop < NHOPS; ++hop) {
    const float*  as  = att_s + (size_t)hop * HDIM;
    const float*  ad  = att_d + (size_t)hop * HDIM;
    const float*  bg  = b_gat + (size_t)hop * HDIM;
    const __bf16* wg  = w_gatT + (size_t)hop * HDIM * HDIM;
    const __bf16* wih = wih_bf + (size_t)hop * 4 * HDIM * 2 * HDIM;
    const __bf16* whh = whh_bf + (size_t)hop * 4 * HDIM * HDIM;
    const __bf16* xcur = (hop == 0) ? h0_bf : hp;   // reference: x_cur = h after each step

    // xp = h0 @ w_gat[hop]   (f32 out for attention/aggregation)
    wmma_gemm_bt<<<wgrid((long)(N / 16) * (HDIM / 64)), TPB, 0, stream>>>(
        h0_bf, wg, nullptr, xp, nullptr, N, HDIM, HDIM, HDIM);

    attn_scores<<<wgrid(N), TPB, 0, stream>>>(xp, as, ad, es, ed, N);
    gat_init<<<egrid(N), TPB, 0, stream>>>(mbuf, den, N);
    hipMemsetAsync(agg, 0, (size_t)N * HDIM * sizeof(float), stream);
    edge_max<<<egrid(E), TPB, 0, stream>>>(src, dst, es, ed, mbuf, E);
    m_fixup<<<egrid(N), TPB, 0, stream>>>(mbuf, N);
    edge_exp<<<egrid(E), TPB, 0, stream>>>(src, dst, es, ed, mbuf, xp, exb, den, E);
    edge_scatter<<<wgrid(E), TPB, 0, stream>>>(src, dst, exb, den, xp, agg, E);
    gat_finish<<<egrid((long)N * HDIM), TPB, 0, stream>>>(agg, bg, htmp_bf, N * HDIM);

    // fused LSTM step: 3 GEMMs + recurrence, no gates buffer
    lstm_fused<<<wgrid((long)(N / 16) * (HDIM / 16)), TPB, 0, stream>>>(
        htmp_bf, xcur, hp, wih, whh, cbuf, hn, N);

    __bf16* t = hp; hp = hn; hn = t;   // hp now holds the freshly written h
  }

  // out = h @ w_out + b_out
  wmma_gemm_bt<<<wgrid((long)(N / 16) * (HDIM / 64)), TPB, 0, stream>>>(
      hp, w_outT, b_out, out, nullptr, N, HDIM, HDIM, HDIM);
}
